// SentenceNet_26379689132256
// MI455X (gfx1250) — compile-verified, matrix operand order checked by reading, not run on hardware
//
#include <hip/hip_runtime.h>
#include <hip/hip_bf16.h>
#include <math.h>

// ---------------- problem constants ----------------
#define H_     8
#define C_     640
#define HC_    5120
#define NS_    8192
#define NW_    65536
#define NSEC_  1024
#define EW_    131072
#define ES_    262144
#define ESEC_  16384
#define WORD_  300
#define WORDP_ 320      // padded to multiple of 32 for WMMA K
#define SENT_  640
#define SECT_  512
#define FFN_   2048

typedef __attribute__((ext_vector_type(16))) __bf16 v16bf;
typedef __attribute__((ext_vector_type(8)))  float  v8f;

// ---------------- helpers ----------------
__device__ __forceinline__ float bf2f(unsigned short u) {
    return __uint_as_float(((unsigned int)u) << 16);
}
__device__ __forceinline__ unsigned short f2bf(float f) {
    unsigned int u = __float_as_uint(f);
    u += 0x7FFFu + ((u >> 16) & 1u);          // round-to-nearest-even
    return (unsigned short)(u >> 16);
}
__device__ __forceinline__ void atomicMaxF(float* addr, float val) {
    int old = __float_as_int(*addr);
    while (__int_as_float(old) < val) {
        int prev = atomicCAS((int*)addr, old, __float_as_int(val));
        if (prev == old) break;
        old = prev;
    }
}

// CDNA5 async global->LDS DMA (no VGPR round-trip, tracked by ASYNCcnt).
// Per-lane: copies 16B from the lane's global address to the lane's LDS address.
__device__ __forceinline__ void asyncLoadB128(unsigned ldsAddr, const void* gaddr) {
    asm volatile("global_load_async_to_lds_b128 %0, %1, off"
                 :: "v"(ldsAddr), "v"(gaddr) : "memory");
}
__device__ __forceinline__ void waitAsync0() {
    asm volatile("s_wait_asynccnt 0x0" ::: "memory");
}

// ---------------- elementwise / prep kernels ----------------
__global__ void k_fill(float* p, float v, size_t n) {
    size_t i = (size_t)blockIdx.x * blockDim.x + threadIdx.x;
    if (i < n) p[i] = v;
}

// src [R,Cs] f32 -> dst [R,Cd] bf16 (zero padded cols)
__global__ void k_cvt_pad(const float* __restrict__ src, int R, int Cs, int Cd,
                          unsigned short* __restrict__ dst) {
    size_t i = (size_t)blockIdx.x * blockDim.x + threadIdx.x;
    size_t tot = (size_t)R * Cd;
    if (i >= tot) return;
    int r = (int)(i / Cd), c = (int)(i % Cd);
    float v = (c < Cs) ? src[(size_t)r * Cs + c] : 0.f;
    dst[i] = f2bf(v);
}

// src [K,N] f32 -> dst [N,Kp] bf16 transposed (zero padded K)
__global__ void k_tr_pad(const float* __restrict__ src, int K, int N, int Kp,
                         unsigned short* __restrict__ dst) {
    size_t i = (size_t)blockIdx.x * blockDim.x + threadIdx.x;
    size_t tot = (size_t)N * Kp;
    if (i >= tot) return;
    int n = (int)(i / Kp), k = (int)(i % Kp);
    float v = (k < K) ? src[(size_t)k * N + n] : 0.f;
    dst[i] = f2bf(v);
}

// fold attention vector into weight: u[k,h] = sum_c W[k, h*C+c] * a[h,c]
__global__ void k_fold(const float* __restrict__ W, const float* __restrict__ a,
                       float* __restrict__ u, int K) {
    int i = blockIdx.x * blockDim.x + threadIdx.x;
    if (i >= K * H_) return;
    int k = i / H_, h = i % H_;
    const float* wr = W + (size_t)k * HC_ + (size_t)h * C_;
    const float* ar = a + (size_t)h * C_;
    float s = 0.f;
    for (int c = 0; c < C_; ++c) s += wr[c] * ar[c];
    u[i] = s;
}

// al[n,h] = sum_k X[n,k] * u[k,h]
__global__ void k_al(const float* __restrict__ X, const float* __restrict__ u,
                     float* __restrict__ al, int N, int K) {
    int i = blockIdx.x * blockDim.x + threadIdx.x;
    if (i >= N * H_) return;
    int n = i / H_, h = i % H_;
    const float* xr = X + (size_t)n * K;
    float s = 0.f;
    for (int k = 0; k < K; ++k) s += xr[k] * u[k * H_ + h];
    al[i] = s;
}

// ---------------- GAT edge kernels ----------------
__global__ void k_edge_max(const int* __restrict__ src, const int* __restrict__ dst,
                           const float* __restrict__ als, const float* __restrict__ ald,
                           float* __restrict__ m, int E) {
    int i = blockIdx.x * blockDim.x + threadIdx.x;
    if (i >= E * H_) return;
    int e = i / H_, h = i % H_;
    int s = src[e], d = dst[e];
    float v = als[s * H_ + h] + ald[d * H_ + h];
    v = (v > 0.f) ? v : 0.2f * v;                 // leaky relu 0.2
    atomicMaxF(&m[d * H_ + h], v);
}

__global__ void k_edge_exp(const int* __restrict__ src, const int* __restrict__ dst,
                           const float* __restrict__ als, const float* __restrict__ ald,
                           const float* __restrict__ m, float* __restrict__ ex,
                           float* __restrict__ den, int E) {
    int i = blockIdx.x * blockDim.x + threadIdx.x;
    if (i >= E * H_) return;
    int e = i / H_, h = i % H_;
    int s = src[e], d = dst[e];
    float v = als[s * H_ + h] + ald[d * H_ + h];
    v = (v > 0.f) ? v : 0.2f * v;
    float x = __expf(v - m[d * H_ + h]);
    ex[i] = x;
    unsafeAtomicAdd(&den[d * H_ + h], x);          // global_atomic_add_f32
}

// one workgroup per edge; scatter hs[src]*alpha into out[dst]
__global__ void k_agg(const int* __restrict__ src, const int* __restrict__ dst,
                      const unsigned short* __restrict__ hs, const float* __restrict__ ex,
                      const float* __restrict__ den, float* __restrict__ out) {
    __shared__ float alpha[H_];
    int e = blockIdx.x;
    int s = src[e], d = dst[e];
    if (threadIdx.x < H_)
        alpha[threadIdx.x] = ex[(size_t)e * H_ + threadIdx.x] /
                             (den[d * H_ + threadIdx.x] + 1e-16f);
    __syncthreads();
    const unsigned short* hr = hs + (size_t)s * HC_;
    float* orow = out + (size_t)d * HC_;
    for (int c = threadIdx.x; c < HC_; c += blockDim.x)
        unsafeAtomicAdd(&orow[c], bf2f(hr[c]) * alpha[c / C_]);
}

// U = elu(agg + bias) -> bf16 into a concat slot (dst pre-offset, row stride)
__global__ void k_elu_emit(const float* __restrict__ agg, const float* __restrict__ b,
                           unsigned short* __restrict__ dst, int dstStride) {
    size_t i = (size_t)blockIdx.x * blockDim.x + threadIdx.x;
    size_t tot = (size_t)NS_ * HC_;
    if (i >= tot) return;
    int n = (int)(i / HC_), c = (int)(i % HC_);
    float v = agg[i] + b[c];
    v = (v > 0.f) ? v : expm1f(v);
    dst[(size_t)n * dstStride + c] = f2bf(v);
}

// Ufused = Z*X + (1-Z)*Y where [X|Y] is a bf16 concat buffer (stride 2*HC)
__global__ void k_combine(const float* __restrict__ Z, const unsigned short* __restrict__ XY,
                          unsigned short* __restrict__ dstB, int dstStride) {
    size_t i = (size_t)blockIdx.x * blockDim.x + threadIdx.x;
    size_t tot = (size_t)NS_ * HC_;
    if (i >= tot) return;
    int n = (int)(i / HC_), c = (int)(i % HC_);
    float z = Z[i];
    float x = bf2f(XY[(size_t)n * (2 * HC_) + c]);
    float y = bf2f(XY[(size_t)n * (2 * HC_) + HC_ + c]);
    dstB[(size_t)n * dstStride + c] = f2bf(z * x + (1.f - z) * y);
}

// ---------------- WMMA GEMM: C = A(bf16,[M,K]) * B^T(bf16,[N,K]) ----------------
// 256 threads = 8 waves, 128x128 block tile, BK=32, async double-buffered LDS.
#define BM 128
#define BN 128
#define BK 32
#define LDSS (BK + 8)     // row pad keeps 16B alignment, breaks bank conflicts

__global__ __launch_bounds__(256)
void k_gemm_bf16(const unsigned short* __restrict__ A, int lda,
                 const unsigned short* __restrict__ B, int ldb, int K,
                 const float* __restrict__ bias, const float* __restrict__ residual,
                 int op,                         // 0=none, 1=sigmoid
                 float* __restrict__ outF, int ldcF,
                 unsigned short* __restrict__ outB, int ldcB) {
    // [buffer][A=0/B=1][BM rows x LDSS] -> 40KB of the 320KB WGP LDS pool
    __shared__ unsigned short tiles[2][2][BM * LDSS];

    const int tid = threadIdx.x;
    const int bx = blockIdx.x, by = blockIdx.y;
    const int w = tid >> 5, lane = tid & 31;
    const int wm = w >> 2, wn = w & 3;       // wave grid 2 (M) x 4 (N)
    const int hi = lane >> 4, r16 = lane & 15;

    const v8f vzero = {0.f, 0.f, 0.f, 0.f, 0.f, 0.f, 0.f, 0.f};
    v8f acc[4][2];
#pragma unroll
    for (int mt = 0; mt < 4; ++mt)
#pragma unroll
        for (int nt = 0; nt < 2; ++nt) acc[mt][nt] = vzero;

    const unsigned short* Arow = A + (size_t)by * BM * lda;
    const unsigned short* Brow = B + (size_t)bx * BN * ldb;

    // issue one K-tile's async DMA: 512 chunks of 8 bf16 per matrix, 2 per thread
    auto issueTile = [&](int kk, int buf) {
#pragma unroll
        for (int i = 0; i < 2; ++i) {
            int c = tid + i * 256;
            int row = c >> 2, cc = (c & 3) * 8;
            asyncLoadB128((unsigned)(size_t)&tiles[buf][0][row * LDSS + cc],
                          &Arow[(size_t)row * lda + kk + cc]);
            asyncLoadB128((unsigned)(size_t)&tiles[buf][1][row * LDSS + cc],
                          &Brow[(size_t)row * ldb + kk + cc]);
        }
    };

    issueTile(0, 0);
    waitAsync0();
    __syncthreads();

    union Frag { uint4 q[2]; v16bf v; };
    int cur = 0;
    for (int kk = 0; kk < K; kk += BK) {
        if (kk + BK < K) issueTile(kk + BK, cur ^ 1);   // prefetch next tile (async)

        const unsigned short* As = tiles[cur][0];
        const unsigned short* Bs = tiles[cur][1];

        // ISA 16-bit A layout: lanes 0-15 hold K {0..7,16..23}, lanes 16-31 +8.
        v16bf aF[4], bF[2];
#pragma unroll
        for (int mt = 0; mt < 4; ++mt) {
            int row = wm * 64 + mt * 16 + r16;
            Frag f;
            f.q[0] = *(const uint4*)&As[row * LDSS + hi * 8];
            f.q[1] = *(const uint4*)&As[row * LDSS + 16 + hi * 8];
            aF[mt] = f.v;
        }
#pragma unroll
        for (int nt = 0; nt < 2; ++nt) {
            int col = wn * 32 + nt * 16 + r16;
            Frag f;
            f.q[0] = *(const uint4*)&Bs[col * LDSS + hi * 8];
            f.q[1] = *(const uint4*)&Bs[col * LDSS + 16 + hi * 8];
            bF[nt] = f.v;
        }
#pragma unroll
        for (int mt = 0; mt < 4; ++mt)
#pragma unroll
            for (int nt = 0; nt < 2; ++nt)
                acc[mt][nt] = __builtin_amdgcn_wmma_f32_16x16x32_bf16(
                    false, aF[mt], false, bF[nt], (short)0, acc[mt][nt], false, false);

        // own async issues done, then publish tile / prove old buffer drained
        waitAsync0();
        __syncthreads();
        cur ^= 1;
    }

    // epilogue: D layout lane l, reg r -> M = base + (l/16)*8 + r, N = base + l%16
    const int colWave = bx * BN + wn * 32;
    const int rowWave = by * BM + wm * 64 + hi * 8;
#pragma unroll
    for (int mt = 0; mt < 4; ++mt) {
#pragma unroll
        for (int nt = 0; nt < 2; ++nt) {
            int col = colWave + nt * 16 + r16;
            float bv = bias ? bias[col] : 0.f;
            int rowBase = rowWave + mt * 16;
#pragma unroll
            for (int rr = 0; rr < 8; ++rr) {
                float v = acc[mt][nt][rr] + bv;
                if (op == 1) v = 1.f / (1.f + __expf(-v));
                int row = rowBase + rr;
                if (residual) v += residual[(size_t)row * ldcF + col];
                if (outF) outF[(size_t)row * ldcF + col] = v;
                if (outB) outB[(size_t)row * ldcB + col] = f2bf(v);
            }
        }
    }
}

// ---------------- host orchestration ----------------
static inline dim3 g1(size_t n) { return dim3((unsigned)((n + 255) / 256)); }

extern "C" void kernel_launch(void* const* d_in, const int* in_sizes, int n_in,
                              void* d_out, int out_size, void* d_ws, size_t ws_size,
                              hipStream_t stream) {
    const float* Hs    = (const float*)d_in[0];
    const float* Hw    = (const float*)d_in[1];
    const float* HS    = (const float*)d_in[2];
    const int*   w2s   = (const int*)d_in[3];
    const int*   s2s   = (const int*)d_in[4];
    const int*   S2s   = (const int*)d_in[5];
    const float* Ww_src = (const float*)d_in[6];
    const float* Ww_dst = (const float*)d_in[7];
    const float* aw_src = (const float*)d_in[8];
    const float* aw_dst = (const float*)d_in[9];
    const float* bw     = (const float*)d_in[10];
    const float* Ws_    = (const float*)d_in[11];
    const float* as_src = (const float*)d_in[12];
    const float* as_dst = (const float*)d_in[13];
    const float* bs     = (const float*)d_in[14];
    const float* WS_src = (const float*)d_in[15];
    const float* WS_dst = (const float*)d_in[16];
    const float* aS_src = (const float*)d_in[17];
    const float* aS_dst = (const float*)d_in[18];
    const float* bS     = (const float*)d_in[19];
    const float* F1_w   = (const float*)d_in[20];
    const float* F1_b   = (const float*)d_in[21];
    const float* F2_w   = (const float*)d_in[22];
    const float* F2_b   = (const float*)d_in[23];
    const float* l1_w   = (const float*)d_in[24];
    const float* l1_b   = (const float*)d_in[25];
    const float* l2_w   = (const float*)d_in[26];
    const float* l2_b   = (const float*)d_in[27];
    float* out = (float*)d_out;

    // bump allocator over workspace (256B aligned)
    char* ws = (char*)d_ws;
    size_t off = 0;
    auto alloc = [&](size_t bytes) -> void* {
        off = (off + 255) & ~(size_t)255;
        void* p = ws + off;
        off += bytes;
        return p;
    };
    typedef unsigned short u16;
    u16* HwP  = (u16*)alloc((size_t)NW_ * WORDP_ * 2);
    u16* HsB  = (u16*)alloc((size_t)NS_ * SENT_ * 2);
    u16* HSB  = (u16*)alloc((size_t)NSEC_ * SECT_ * 2);
    u16* WwsT = (u16*)alloc((size_t)HC_ * WORDP_ * 2);
    u16* WsT  = (u16*)alloc((size_t)HC_ * SENT_ * 2);
    u16* WSsT = (u16*)alloc((size_t)HC_ * SECT_ * 2);
    u16* F1T  = (u16*)alloc((size_t)HC_ * (2 * HC_) * 2);
    u16* F2T  = (u16*)alloc((size_t)HC_ * (2 * HC_) * 2);
    u16* l1T  = (u16*)alloc((size_t)FFN_ * HC_ * 2);
    u16* l2T  = (u16*)alloc((size_t)SENT_ * FFN_ * 2);
    float* uws = (float*)alloc(WORD_ * H_ * 4);
    float* vwd = (float*)alloc(SENT_ * H_ * 4);
    float* uss = (float*)alloc(SENT_ * H_ * 4);
    float* usd = (float*)alloc(SENT_ * H_ * 4);
    float* uSs = (float*)alloc(SECT_ * H_ * 4);
    float* vSd = (float*)alloc(SENT_ * H_ * 4);
    float* alw_s = (float*)alloc((size_t)NW_ * H_ * 4);
    float* alw_d = (float*)alloc((size_t)NS_ * H_ * 4);
    float* als_s = (float*)alloc((size_t)NS_ * H_ * 4);
    float* als_d = (float*)alloc((size_t)NS_ * H_ * 4);
    float* alS_s = (float*)alloc((size_t)NSEC_ * H_ * 4);
    float* alS_d = (float*)alloc((size_t)NS_ * H_ * 4);
    u16* hswB = (u16*)alloc((size_t)NW_ * HC_ * 2);
    u16* hssB = (u16*)alloc((size_t)NS_ * HC_ * 2);
    u16* hsSB = (u16*)alloc((size_t)NSEC_ * HC_ * 2);
    float* mbuf = (float*)alloc((size_t)NS_ * H_ * 4);
    float* dbuf = (float*)alloc((size_t)NS_ * H_ * 4);
    float* exbuf = (float*)alloc((size_t)ES_ * H_ * 4);   // sized for largest E
    float* agg  = (float*)alloc((size_t)NS_ * HC_ * 4);   // reused per GAT
    u16* cat1 = (u16*)alloc((size_t)NS_ * (2 * HC_) * 2);
    u16* cat2 = (u16*)alloc((size_t)NS_ * (2 * HC_) * 2);
    float* Zpre = (float*)alloc((size_t)NS_ * HC_ * 4);   // reused for both gates
    u16* U2B  = (u16*)alloc((size_t)NS_ * HC_ * 2);
    u16* Y1B  = (u16*)alloc((size_t)NS_ * FFN_ * 2);
    (void)ws_size; (void)n_in; (void)in_sizes; (void)out_size;

    // --- conversions (fp32 -> bf16, weights transposed to [N,K]) ---
    k_cvt_pad<<<g1((size_t)NW_ * WORDP_), 256, 0, stream>>>(Hw, NW_, WORD_, WORDP_, HwP);
    k_cvt_pad<<<g1((size_t)NS_ * SENT_), 256, 0, stream>>>(Hs, NS_, SENT_, SENT_, HsB);
    k_cvt_pad<<<g1((size_t)NSEC_ * SECT_), 256, 0, stream>>>(HS, NSEC_, SECT_, SECT_, HSB);
    k_tr_pad<<<g1((size_t)HC_ * WORDP_), 256, 0, stream>>>(Ww_src, WORD_, HC_, WORDP_, WwsT);
    k_tr_pad<<<g1((size_t)HC_ * SENT_), 256, 0, stream>>>(Ws_, SENT_, HC_, SENT_, WsT);
    k_tr_pad<<<g1((size_t)HC_ * SECT_), 256, 0, stream>>>(WS_src, SECT_, HC_, SECT_, WSsT);
    k_tr_pad<<<g1((size_t)HC_ * 2 * HC_), 256, 0, stream>>>(F1_w, 2 * HC_, HC_, 2 * HC_, F1T);
    k_tr_pad<<<g1((size_t)HC_ * 2 * HC_), 256, 0, stream>>>(F2_w, 2 * HC_, HC_, 2 * HC_, F2T);
    k_tr_pad<<<g1((size_t)FFN_ * HC_), 256, 0, stream>>>(l1_w, HC_, FFN_, HC_, l1T);
    k_tr_pad<<<g1((size_t)SENT_ * FFN_), 256, 0, stream>>>(l2_w, FFN_, SENT_, FFN_, l2T);

    // --- fold attention vectors (dst projections never materialized) ---
    k_fold<<<g1(WORD_ * H_), 256, 0, stream>>>(Ww_src, aw_src, uws, WORD_);
    k_fold<<<g1(SENT_ * H_), 256, 0, stream>>>(Ww_dst, aw_dst, vwd, SENT_);
    k_fold<<<g1(SENT_ * H_), 256, 0, stream>>>(Ws_, as_src, uss, SENT_);
    k_fold<<<g1(SENT_ * H_), 256, 0, stream>>>(Ws_, as_dst, usd, SENT_);
    k_fold<<<g1(SECT_ * H_), 256, 0, stream>>>(WS_src, aS_src, uSs, SECT_);
    k_fold<<<g1(SENT_ * H_), 256, 0, stream>>>(WS_dst, aS_dst, vSd, SENT_);
    k_al<<<g1((size_t)NW_ * H_), 256, 0, stream>>>(Hw, uws, alw_s, NW_, WORD_);
    k_al<<<g1((size_t)NS_ * H_), 256, 0, stream>>>(Hs, vwd, alw_d, NS_, SENT_);
    k_al<<<g1((size_t)NS_ * H_), 256, 0, stream>>>(Hs, uss, als_s, NS_, SENT_);
    k_al<<<g1((size_t)NS_ * H_), 256, 0, stream>>>(Hs, usd, als_d, NS_, SENT_);
    k_al<<<g1((size_t)NSEC_ * H_), 256, 0, stream>>>(HS, uSs, alS_s, NSEC_, SECT_);
    k_al<<<g1((size_t)NS_ * H_), 256, 0, stream>>>(Hs, vSd, alS_d, NS_, SENT_);

    // --- source projections (WMMA) ---
    k_gemm_bf16<<<dim3(HC_ / BN, NW_ / BM), 256, 0, stream>>>(
        HwP, WORDP_, WwsT, WORDP_, WORDP_, nullptr, nullptr, 0, nullptr, 0, hswB, HC_);
    k_gemm_bf16<<<dim3(HC_ / BN, NS_ / BM), 256, 0, stream>>>(
        HsB, SENT_, WsT, SENT_, SENT_, nullptr, nullptr, 0, nullptr, 0, hssB, HC_);
    k_gemm_bf16<<<dim3(HC_ / BN, NSEC_ / BM), 256, 0, stream>>>(
        HSB, SECT_, WSsT, SECT_, SECT_, nullptr, nullptr, 0, nullptr, 0, hsSB, HC_);

    // --- three GATs: softmax over incoming edges + scatter, ELU, emit bf16 ---
    struct Gat {
        const int* ei; int E; const float* als; const float* ald;
        const u16* hs; const float* bias; u16* dst;
    } gats[3] = {
        { w2s, EW_,   alw_s, alw_d, hswB, bw, cat1 },          // word -> sent (U_w)
        { s2s, ES_,   als_s, als_d, hssB, bs, cat1 + HC_ },    // sent -> sent (U_s)
        { S2s, ESEC_, alS_s, alS_d, hsSB, bS, cat2 + HC_ },    // sect -> sent (U_S)
    };
    for (int g = 0; g < 3; ++g) {
        const int* src = gats[g].ei;
        const int* dst = gats[g].ei + gats[g].E;
        k_fill<<<g1((size_t)NS_ * H_), 256, 0, stream>>>(mbuf, -3.0e38f, (size_t)NS_ * H_);
        k_fill<<<g1((size_t)NS_ * H_), 256, 0, stream>>>(dbuf, 0.f, (size_t)NS_ * H_);
        k_fill<<<g1((size_t)NS_ * HC_), 256, 0, stream>>>(agg, 0.f, (size_t)NS_ * HC_);
        k_edge_max<<<g1((size_t)gats[g].E * H_), 256, 0, stream>>>(
            src, dst, gats[g].als, gats[g].ald, mbuf, gats[g].E);
        k_edge_exp<<<g1((size_t)gats[g].E * H_), 256, 0, stream>>>(
            src, dst, gats[g].als, gats[g].ald, mbuf, exbuf, dbuf, gats[g].E);
        k_agg<<<dim3(gats[g].E), 256, 0, stream>>>(src, dst, gats[g].hs, exbuf, dbuf, agg);
        k_elu_emit<<<g1((size_t)NS_ * HC_), 256, 0, stream>>>(
            agg, gats[g].bias, gats[g].dst, 2 * HC_);
    }

    // --- fusion gate 1: Z = sigmoid([Uw|Us]@F1 + b); U1 = Z*Uw + (1-Z)*Us ---
    k_gemm_bf16<<<dim3(HC_ / BN, NS_ / BM), 256, 0, stream>>>(
        cat1, 2 * HC_, F1T, 2 * HC_, 2 * HC_, F1_b, nullptr, 1, Zpre, HC_, nullptr, 0);
    k_combine<<<g1((size_t)NS_ * HC_), 256, 0, stream>>>(Zpre, cat1, cat2, 2 * HC_);

    // --- fusion gate 2: U2 = Z2*U1 + (1-Z2)*US ---
    k_gemm_bf16<<<dim3(HC_ / BN, NS_ / BM), 256, 0, stream>>>(
        cat2, 2 * HC_, F2T, 2 * HC_, 2 * HC_, F2_b, nullptr, 1, Zpre, HC_, nullptr, 0);
    k_combine<<<g1((size_t)NS_ * HC_), 256, 0, stream>>>(Zpre, cat2, U2B, HC_);

    // --- FFN: (U2@l1 + b1)@l2 + b2 + Hs ---
    k_gemm_bf16<<<dim3(FFN_ / BN, NS_ / BM), 256, 0, stream>>>(
        U2B, HC_, l1T, HC_, HC_, l1_b, nullptr, 0, nullptr, 0, Y1B, FFN_);
    k_gemm_bf16<<<dim3(SENT_ / BN, NS_ / BM), 256, 0, stream>>>(
        Y1B, FFN_, l2T, FFN_, FFN_, l2_b, Hs, 0, out, SENT_, nullptr, 0);
}
